// Decoder_60498909331476
// MI455X (gfx1250) — matrix operand order repaired and checked by reading, not compiled, not run on hardware
//
#include <hip/hip_runtime.h>
#include <hip/hip_bf16.h>

// ---------------------------------------------------------------------------
// Decoder forward for MI455X (gfx1250, wave32, WMMA + Tensor Data Mover).
// GEMMs run on v_wmma_f32_16x16x32_bf16; large GEMMs stage operands into LDS
// with tensor_load_to_lds (TDM) double buffering, waited via s_wait_tensorcnt.
// ---------------------------------------------------------------------------

constexpr int V_   = 32000;
constexpr int D_   = 1024;
constexpr int DI_  = 768;
constexpr int S_   = 512;
constexpr int B_   = 8;
constexpr int NI_  = 197;
constexpr int NIP_ = 224;   // NI padded up to a multiple of 32 (WMMA K step)

typedef __attribute__((ext_vector_type(16))) __bf16 bf16x16;
typedef __attribute__((ext_vector_type(8)))  float  f32x8;
typedef unsigned int u32x4 __attribute__((ext_vector_type(4)));
typedef int          i32x8 __attribute__((ext_vector_type(8)));
typedef int          i32x4 __attribute__((ext_vector_type(4)));

union BfFrag { bf16x16 v; uint4 q[2]; };

__device__ __forceinline__ float wave_max(float v) {
  for (int o = 16; o; o >>= 1) v = fmaxf(v, __shfl_xor(v, o, 32));
  return v;
}
__device__ __forceinline__ float wave_sum(float v) {
  for (int o = 16; o; o >>= 1) v += __shfl_xor(v, o, 32);
  return v;
}

// ---------------------------------------------------------------------------
// TDM: issue an async 2-D bf16 tile load (tile1 rows x tile0 cols, row stride
// `stride` elements) from global into LDS at byte offset lds_addr.
// D# layout per CDNA5 ISA ch.8 (group0: count/lds/global/type, group1: dims).
// Completion tracked on TENSORcnt (one "done" per instruction).
// Toolchain: 6-arg clang-23 builtin (g0, g1, g2, g3, g4, cpol).
// ---------------------------------------------------------------------------
__device__ __forceinline__ void tdm_load_tile(unsigned lds_addr, const void* gptr,
                                              unsigned tile0, unsigned tile1,
                                              unsigned long long stride_elems) {
  unsigned long long ga = (unsigned long long)gptr;
  u32x4 g0;
  g0[0] = 1u;                                                  // count=1, user D#
  g0[1] = lds_addr;                                            // LDS byte address
  g0[2] = (unsigned)ga;                                        // global_addr[31:0]
  g0[3] = (unsigned)((ga >> 32) & 0x01FFFFFFu) | (2u << 30);   // addr[56:32] | type=2
  i32x8 g1;
  g1[0] = (int)(1u << 16);                        // data_size = 2 bytes; no pad/iter/mcast
  g1[1] = (int)((tile0 & 0xFFFFu) << 16);         // tensor_dim0[15:0] (== tile0)
  g1[2] = (int)((tile1 & 0xFFFFu) << 16);         // tensor_dim0[31:16]=0 | tensor_dim1[15:0]
  g1[3] = (int)((tile0 & 0xFFFFu) << 16);         // tensor_dim1[31:16]=0 | tile_dim0
  g1[4] = (int)(tile1 & 0xFFFFu);                 // tile_dim1 | tile_dim2=0
  g1[5] = (int)(unsigned)(stride_elems & 0xFFFFFFFFu);      // dim0 stride [31:0]
  g1[6] = (int)(unsigned)((stride_elems >> 32) & 0xFFFFu);  // stride[47:32] | dim1_stride=0
  g1[7] = 0;
  i32x4 z4 = {0, 0, 0, 0};                        // groups 2/3 unused (2-D tensor)
  i32x8 z8 = {0, 0, 0, 0, 0, 0, 0, 0};
  __builtin_amdgcn_tensor_load_to_lds(g0, g1, z4, z4, z8, 0);
}

// ---------------------------------------------------------------------------
// f32 -> bf16 conversion (weights / image embeddings)
// ---------------------------------------------------------------------------
__global__ void cvt_bf16_kernel(const float* __restrict__ in,
                                __bf16* __restrict__ out, long n) {
  long i = (long)blockIdx.x * blockDim.x + threadIdx.x;
  if (i < n) out[i] = (__bf16)in[i];
}

// ---------------------------------------------------------------------------
// x = emb_table[tokens] + sinusoidal positional encoding; also bf16 copy
// ---------------------------------------------------------------------------
__global__ __launch_bounds__(256)
void embed_kernel(const int* __restrict__ tokens, const float* __restrict__ emb,
                  float* __restrict__ x32, __bf16* __restrict__ xb) {
  int row = blockIdx.x;                    // row in [0, B*S)
  int s   = row % S_;
  int tok = tokens[row];
  const float* e = emb + (size_t)tok * D_;
  float*  xo  = x32 + (size_t)row * D_;
  __bf16* xbo = xb  + (size_t)row * D_;
  for (int d = threadIdx.x; d < D_; d += 256) {
    int i = d & ~1;
    float ang = (float)s * __powf(10000.0f, -(float)i / (float)D_);
    float pe  = (d & 1) ? __cosf(ang) : __sinf(ang);
    float v   = e[d] + pe;
    xo[d]  = v;
    xbo[d] = (__bf16)v;
  }
}

// ---------------------------------------------------------------------------
// TDM-staged WMMA GEMM:  C[b] = scale * (A[b] x W[b]) + bias
// Requires: M % 128 == 0, N % 128 == 0, K % 32 == 0.
// Block: 256 threads = 8 waves; block tile 128x128; wave tile 32x64 (2x4 frags).
// Wave 0 drives the Tensor Data Mover: per K-step it issues 2-D tile loads
// (A: 128x32, B: 32x128) into the ping-pong LDS buffers, s_wait_tensorcnt 0,
// and the workgroup barrier publishes the slab to all 8 waves.
// ---------------------------------------------------------------------------
template<bool HAS_BIAS, bool OUT_BF16>
__global__ __launch_bounds__(256)
void gemm_wmma_tdm(const __bf16* __restrict__ A, const __bf16* __restrict__ W,
                   const float* __restrict__ bias,
                   float* __restrict__ Cf, __bf16* __restrict__ Cb,
                   int M, int N, int K, int lda, int ldw, int ldc,
                   long sA, long sW, long sC, float scale) {
  __shared__ __align__(16) __bf16 sAt[2][128 * 32];   // [row][k] packed, 8 KB each
  __shared__ __align__(16) __bf16 sBt[2][32 * 128];   // [k][col] packed, 8 KB each

  const int bz = blockIdx.z;
  A += (size_t)bz * sA;
  W += (size_t)bz * sW;
  if (Cf) Cf += (size_t)bz * sC;
  if (Cb) Cb += (size_t)bz * sC;

  const int mblk = blockIdx.y * 128;
  const int nblk = blockIdx.x * 128;
  const int lane = threadIdx.x & 31;
  const int wv   = threadIdx.x >> 5;
  const int m0w  = (wv >> 1) * 32;       // 4 wave-rows x 2 wave-cols
  const int n0w  = (wv & 1) * 64;
  const int l15  = lane & 15;
  const int kb   = (lane < 16) ? 0 : 8;  // A-fragment K sub-block offset

  f32x8 acc[2][4] = {};
  const int nsteps = K >> 5;

  if (wv == 0) {
    tdm_load_tile((unsigned)(uintptr_t)&sAt[0][0], A + (size_t)mblk * lda,
                  32u, 128u, (unsigned long long)lda);
    tdm_load_tile((unsigned)(uintptr_t)&sBt[0][0], W + nblk,
                  128u, 32u, (unsigned long long)ldw);
  }

  int cur = 0;
  for (int ks = 0; ks < nsteps; ks++) {
    if (wv == 0) __builtin_amdgcn_s_wait_tensorcnt((short)0);
    __syncthreads();                       // slab `cur` visible to all waves
    if (wv == 0 && ks + 1 < nsteps) {      // prefetch next slab into other buffer
      int k0n = (ks + 1) << 5;
      tdm_load_tile((unsigned)(uintptr_t)&sAt[cur ^ 1][0],
                    A + (size_t)mblk * lda + k0n, 32u, 128u, (unsigned long long)lda);
      tdm_load_tile((unsigned)(uintptr_t)&sBt[cur ^ 1][0],
                    W + (size_t)k0n * ldw + nblk, 128u, 32u, (unsigned long long)ldw);
    }
    const __bf16* As = &sAt[cur][0];
    const __bf16* Bs = &sBt[cur][0];
    BfFrag a[2], b[4];
#pragma unroll
    for (int i = 0; i < 2; i++) {
      const __bf16* base = As + (size_t)(m0w + i * 16 + l15) * 32 + kb;
      a[i].q[0] = *(const uint4*)(base);        // K = kb .. kb+7
      a[i].q[1] = *(const uint4*)(base + 16);   // K = kb+16 .. kb+23
    }
#pragma unroll
    for (int j = 0; j < 4; j++) {
      for (int e = 0; e < 16; e++)
        b[j].v[e] = Bs[(size_t)(kb + (e < 8 ? e : e + 8)) * 128 + n0w + j * 16 + l15];
    }
#pragma unroll
    for (int i = 0; i < 2; i++)
#pragma unroll
      for (int j = 0; j < 4; j++)
        acc[i][j] = __builtin_amdgcn_wmma_f32_16x16x32_bf16(
            false, a[i].v, false, b[j].v, (short)0, acc[i][j], false, false);
    cur ^= 1;
  }

  const int moff = (lane < 16) ? 0 : 8;
#pragma unroll
  for (int i = 0; i < 2; i++) {
#pragma unroll
    for (int j = 0; j < 4; j++) {
      int n = nblk + n0w + j * 16 + l15;
      float bv = HAS_BIAS ? bias[n] : 0.0f;
#pragma unroll
      for (int r = 0; r < 8; r++) {
        int m = mblk + m0w + i * 16 + moff + r;
        float v = acc[i][j][r] * scale + bv;
        if (OUT_BF16) Cb[(size_t)m * ldc + n] = (__bf16)v;
        else          Cf[(size_t)m * ldc + n] = v;
      }
    }
  }
}

// ---------------------------------------------------------------------------
// Generic batched WMMA GEMM (direct-from-global fragments), used for the
// transposed-B score GEMMs and the ragged-M (1576) K2/V2 projections.
//   TRANSB: logical W[k][n] = Wmem[n*ldw + k]
//   CAUSAL: skip 64x64 blocks entirely above the diagonal.
// Block = 128 threads = 4 waves; block tile 64x64; wave tile 32x32.
// ---------------------------------------------------------------------------
template<bool HAS_BIAS, bool OUT_BF16, bool TRANSB, bool CAUSAL>
__global__ __launch_bounds__(128)
void gemm_wmma(const __bf16* __restrict__ A, const __bf16* __restrict__ W,
               const float* __restrict__ bias,
               float* __restrict__ Cf, __bf16* __restrict__ Cb,
               int M, int N, int K, int lda, int ldw, int ldc,
               long sA, long sW, long sC, float scale,
               int nclamp, int nstore) {
  if (CAUSAL) {
    int mb = blockIdx.y * 64, nb = blockIdx.x * 64;
    if (nb > mb + 63) return;
  }
  int bz = blockIdx.z;
  A += (size_t)bz * sA;
  W += (size_t)bz * sW;
  if (Cf) Cf += (size_t)bz * sC;
  if (Cb) Cb += (size_t)bz * sC;

  const int lane = threadIdx.x & 31;
  const int wv   = threadIdx.x >> 5;
  const int m0   = blockIdx.y * 64 + (wv >> 1) * 32;
  const int n0   = blockIdx.x * 64 + (wv & 1) * 32;
  const int l15  = lane & 15;
  const int kb   = (lane < 16) ? 0 : 8;

  f32x8 acc[2][2] = {};
  int arow[2];
#pragma unroll
  for (int i = 0; i < 2; i++) arow[i] = min(m0 + i * 16 + l15, M - 1);

  for (int k0 = 0; k0 < K; k0 += 32) {
    BfFrag a[2], b[2];
#pragma unroll
    for (int i = 0; i < 2; i++) {
      const __bf16* base = A + (size_t)arow[i] * lda + k0 + kb;
      a[i].q[0] = *(const uint4*)(base);
      a[i].q[1] = *(const uint4*)(base + 16);
    }
    const int krow = k0 + lane;
    if (!TRANSB) {
#pragma unroll
      for (int j = 0; j < 2; j++) {
        for (int e = 0; e < 16; e++)
          b[j].v[e] = W[(size_t)(k0 + kb + (e < 8 ? e : e + 8)) * ldw + n0 + j * 16 + l15];
      }
    } else {
#pragma unroll
      for (int j = 0; j < 2; j++) {
#pragma unroll
        for (int e = 0; e < 16; e++) {
          int nn = min(n0 + j * 16 + l15, nclamp - 1);
          b[j].v[e] = W[(size_t)nn * ldw + k0 + kb + (e < 8 ? e : e + 8)];
        }
      }
    }
#pragma unroll
    for (int i = 0; i < 2; i++)
#pragma unroll
      for (int j = 0; j < 2; j++)
        acc[i][j] = __builtin_amdgcn_wmma_f32_16x16x32_bf16(
            false, a[i].v, false, b[j].v, (short)0, acc[i][j], false, false);
  }

  const int moff = (lane < 16) ? 0 : 8;
#pragma unroll
  for (int i = 0; i < 2; i++) {
#pragma unroll
    for (int j = 0; j < 2; j++) {
      int n = n0 + j * 16 + l15;
      if (n >= nstore) continue;
      float bv = HAS_BIAS ? bias[n] : 0.0f;
#pragma unroll
      for (int r = 0; r < 8; r++) {
        int m = m0 + i * 16 + moff + r;
        if (m >= M) continue;
        float v = acc[i][j][r] * scale + bv;
        if (OUT_BF16) Cb[(size_t)m * ldc + n] = (__bf16)v;
        else          Cf[(size_t)m * ldc + n] = v;
      }
    }
  }
}

// ---------------------------------------------------------------------------
// Causal softmax over scores[B*S, S]; writes bf16 probs, zeros for k > q.
// ---------------------------------------------------------------------------
__global__ __launch_bounds__(128)
void softmax_causal_kernel(const float* __restrict__ sc,
                           __bf16* __restrict__ pb) {
  int row = blockIdx.x;
  int q   = row % S_;
  const float* r = sc + (size_t)row * S_;
  __bf16*      o = pb + (size_t)row * S_;
  __shared__ float redm[4], reds[4];
  int lane = threadIdx.x & 31, wv = threadIdx.x >> 5;

  float mx = -3.4e38f;
  for (int k = threadIdx.x; k <= q; k += 128) mx = fmaxf(mx, r[k]);
  mx = wave_max(mx);
  if (lane == 0) redm[wv] = mx;
  __syncthreads();
  mx = fmaxf(fmaxf(redm[0], redm[1]), fmaxf(redm[2], redm[3]));

  float sum = 0.0f;
  for (int k = threadIdx.x; k <= q; k += 128) sum += __expf(r[k] - mx);
  sum = wave_sum(sum);
  if (lane == 0) reds[wv] = sum;
  __syncthreads();
  sum = reds[0] + reds[1] + reds[2] + reds[3];
  float inv = 1.0f / sum;

  for (int k = threadIdx.x; k < S_; k += 128)
    o[k] = (k <= q) ? (__bf16)(__expf(r[k] - mx) * inv) : (__bf16)0.0f;
}

// ---------------------------------------------------------------------------
// Cross-attn softmax over scores[B*S, NIP_]; zero pads cols [NI_, NIP_).
// ---------------------------------------------------------------------------
__global__ __launch_bounds__(128)
void softmax_cross_kernel(const float* __restrict__ sc,
                          __bf16* __restrict__ pb) {
  int row = blockIdx.x;
  const float* r = sc + (size_t)row * NIP_;
  __bf16*      o = pb + (size_t)row * NIP_;
  __shared__ float redm[4], reds[4];
  int lane = threadIdx.x & 31, wv = threadIdx.x >> 5;

  float mx = -3.4e38f;
  for (int k = threadIdx.x; k < NI_; k += 128) mx = fmaxf(mx, r[k]);
  mx = wave_max(mx);
  if (lane == 0) redm[wv] = mx;
  __syncthreads();
  mx = fmaxf(fmaxf(redm[0], redm[1]), fmaxf(redm[2], redm[3]));

  float sum = 0.0f;
  for (int k = threadIdx.x; k < NI_; k += 128) sum += __expf(r[k] - mx);
  sum = wave_sum(sum);
  if (lane == 0) reds[wv] = sum;
  __syncthreads();
  sum = reds[0] + reds[1] + reds[2] + reds[3];
  float inv = 1.0f / sum;

  for (int k = threadIdx.x; k < NIP_; k += 128)
    o[k] = (k < NI_) ? (__bf16)(__expf(r[k] - mx) * inv) : (__bf16)0.0f;
}

// ---------------------------------------------------------------------------
// y = LayerNorm(a + x) * g + b  -> f32 and bf16 outputs. One block per row.
// ---------------------------------------------------------------------------
__global__ __launch_bounds__(256)
void add_ln_kernel(const float* __restrict__ a, const float* __restrict__ x,
                   const float* __restrict__ g, const float* __restrict__ bt,
                   float* __restrict__ yf, __bf16* __restrict__ yb) {
  int row = blockIdx.x;
  const float* ar = a + (size_t)row * D_;
  const float* xr = x + (size_t)row * D_;
  __shared__ float rs[8], rq[8];
  int lane = threadIdx.x & 31, wv = threadIdx.x >> 5;
  int d0 = threadIdx.x * 4;

  float v[4];
  float s = 0.0f, s2 = 0.0f;
#pragma unroll
  for (int j = 0; j < 4; j++) {
    float t = ar[d0 + j] + xr[d0 + j];
    v[j] = t; s += t; s2 += t * t;
  }
  s = wave_sum(s); s2 = wave_sum(s2);
  if (lane == 0) { rs[wv] = s; rq[wv] = s2; }
  __syncthreads();
  float S1 = 0.0f, S2 = 0.0f;
#pragma unroll
  for (int w = 0; w < 8; w++) { S1 += rs[w]; S2 += rq[w]; }
  float mean = S1 / (float)D_;
  float var  = S2 / (float)D_ - mean * mean;
  float rstd = rsqrtf(var + 1e-5f);
#pragma unroll
  for (int j = 0; j < 4; j++) {
    int d = d0 + j;
    float t = (v[j] - mean) * rstd * g[d] + bt[d];
    yf[(size_t)row * D_ + d] = t;
    yb[(size_t)row * D_ + d] = (__bf16)t;
  }
}

// ---------------------------------------------------------------------------
// Host launcher
// ---------------------------------------------------------------------------
extern "C" void kernel_launch(void* const* d_in, const int* in_sizes, int n_in,
                              void* d_out, int out_size, void* d_ws, size_t ws_size,
                              hipStream_t stream) {
  (void)in_sizes; (void)n_in; (void)out_size; (void)ws_size;
  const int*   tokens = (const int*)  d_in[0];
  const float* img    = (const float*)d_in[1];
  const float* emb    = (const float*)d_in[2];
  const float* Wq1 = (const float*)d_in[3],  *bq1 = (const float*)d_in[4];
  const float* Wk1 = (const float*)d_in[5],  *bk1 = (const float*)d_in[6];
  const float* Wv1 = (const float*)d_in[7],  *bv1 = (const float*)d_in[8];
  const float* g1  = (const float*)d_in[9],  *b1  = (const float*)d_in[10];
  const float* Wq2 = (const float*)d_in[11], *bq2 = (const float*)d_in[12];
  const float* Wk2 = (const float*)d_in[13], *bk2 = (const float*)d_in[14];
  const float* Wv2 = (const float*)d_in[15], *bv2 = (const float*)d_in[16];
  const float* g2  = (const float*)d_in[17], *b2  = (const float*)d_in[18];
  const float* Wp  = (const float*)d_in[19], *bp  = (const float*)d_in[20];
  float* out = (float*)d_out;

  // ---- workspace layout (~190 MB, 256B aligned slabs) --------------------
  size_t off = 0;
  auto alloc = [&](size_t elems, size_t esz) {
    void* p = (char*)d_ws + off;
    off += (elems * esz + 255) & ~(size_t)255;
    return p;
  };
  const size_t MS = (size_t)B_ * S_;     // 4096 activation rows
  float*  x32     = (float*) alloc(MS * D_, 4);
  float*  attn32  = (float*) alloc(MS * D_, 4);
  float*  x2f     = (float*) alloc(MS * D_, 4);
  float*  scoresF = (float*) alloc(MS * S_, 4);
  float*  s2F     = (float*) alloc(MS * NIP_, 4);
  __bf16* xb      = (__bf16*)alloc(MS * D_, 2);   // reused as x3 (final act)
  __bf16* x2b     = (__bf16*)alloc(MS * D_, 2);
  __bf16* imgb    = (__bf16*)alloc((size_t)B_ * NI_ * DI_, 2);
  __bf16* Qb      = (__bf16*)alloc(MS * D_, 2);   // reused as Q2
  __bf16* Kb      = (__bf16*)alloc(MS * D_, 2);   // reused as K2
  __bf16* Vb      = (__bf16*)alloc(MS * D_, 2);   // reused as V2 (+K-pad slack)
  __bf16* probsb  = (__bf16*)alloc(MS * S_, 2);
  __bf16* probs2b = (__bf16*)alloc(MS * NIP_, 2);
  __bf16* Wq1b = (__bf16*)alloc((size_t)D_ * D_, 2);
  __bf16* Wk1b = (__bf16*)alloc((size_t)D_ * D_, 2);
  __bf16* Wv1b = (__bf16*)alloc((size_t)D_ * D_, 2);
  __bf16* Wq2b = (__bf16*)alloc((size_t)D_ * D_, 2);
  __bf16* Wk2b = (__bf16*)alloc((size_t)DI_ * D_, 2);
  __bf16* Wv2b = (__bf16*)alloc((size_t)DI_ * D_, 2);
  __bf16* Wpb  = (__bf16*)alloc((size_t)D_ * V_, 2);

  const dim3 blk128(128), blk256(256);
  const float iscale = 0.03125f;   // 1/sqrt(1024)

  // ---- bf16 conversions ---------------------------------------------------
  auto cvt = [&](const float* s, __bf16* d, long n) {
    cvt_bf16_kernel<<<dim3((unsigned)((n + 255) / 256)), dim3(256), 0, stream>>>(s, d, n);
  };
  cvt(Wq1, Wq1b, (long)D_ * D_);
  cvt(Wk1, Wk1b, (long)D_ * D_);
  cvt(Wv1, Wv1b, (long)D_ * D_);
  cvt(Wq2, Wq2b, (long)D_ * D_);
  cvt(Wk2, Wk2b, (long)DI_ * D_);
  cvt(Wv2, Wv2b, (long)DI_ * D_);
  cvt(Wp,  Wpb,  (long)D_ * V_);
  cvt(img, imgb, (long)B_ * NI_ * DI_);

  // ---- embedding + positional encoding -----------------------------------
  embed_kernel<<<dim3((unsigned)MS), blk256, 0, stream>>>(tokens, emb, x32, xb);

  // ---- self-attention: Q/K/V projections (TDM-staged, bf16 out) -----------
  {
    dim3 g(D_ / 128, (unsigned)(MS / 128), 1);
    gemm_wmma_tdm<true, true><<<g, blk256, 0, stream>>>(
        xb, Wq1b, bq1, nullptr, Qb, (int)MS, D_, D_, D_, D_, D_, 0, 0, 0, 1.0f);
    gemm_wmma_tdm<true, true><<<g, blk256, 0, stream>>>(
        xb, Wk1b, bk1, nullptr, Kb, (int)MS, D_, D_, D_, D_, D_, 0, 0, 0, 1.0f);
    gemm_wmma_tdm<true, true><<<g, blk256, 0, stream>>>(
        xb, Wv1b, bv1, nullptr, Vb, (int)MS, D_, D_, D_, D_, D_, 0, 0, 0, 1.0f);
  }
  // scores = scale * Q K^T (causal blocks above diagonal skipped)
  {
    dim3 g(S_ / 64, S_ / 64, B_);
    gemm_wmma<false, false, true, true><<<g, blk128, 0, stream>>>(
        Qb, Kb, nullptr, scoresF, nullptr, S_, S_, D_, D_, D_, S_,
        (long)S_ * D_, (long)S_ * D_, (long)S_ * S_, iscale, S_, S_);
  }
  softmax_causal_kernel<<<dim3((unsigned)MS), blk128, 0, stream>>>(scoresF, probsb);
  // attn = probs x V   (TDM-staged, batched)
  {
    dim3 g(D_ / 128, S_ / 128, B_);
    gemm_wmma_tdm<false, false><<<g, blk256, 0, stream>>>(
        probsb, Vb, nullptr, attn32, nullptr, S_, D_, S_, S_, D_, D_,
        (long)S_ * S_, (long)S_ * D_, (long)S_ * D_, 1.0f);
  }
  add_ln_kernel<<<dim3((unsigned)MS), blk256, 0, stream>>>(attn32, x32, g1, b1, x2f, x2b);

  // ---- cross-attention ----------------------------------------------------
  {
    dim3 g(D_ / 128, (unsigned)(MS / 128), 1);
    gemm_wmma_tdm<true, true><<<g, blk256, 0, stream>>>(
        x2b, Wq2b, bq2, nullptr, Qb, (int)MS, D_, D_, D_, D_, D_, 0, 0, 0, 1.0f);
  }
  {
    int MI = B_ * NI_;                           // 1576 image-token rows (ragged)
    dim3 g(D_ / 64, (unsigned)((MI + 63) / 64), 1);
    gemm_wmma<true, true, false, false><<<g, blk128, 0, stream>>>(
        imgb, Wk2b, bk2, nullptr, Kb, MI, D_, DI_, DI_, D_, D_, 0, 0, 0, 1.0f, D_, D_);
    gemm_wmma<true, true, false, false><<<g, blk128, 0, stream>>>(
        imgb, Wv2b, bv2, nullptr, Vb, MI, D_, DI_, DI_, D_, D_, 0, 0, 0, 1.0f, D_, D_);
  }
  // scores2 = scale * Q2 K2^T  (N=197, stored with ld=224; loads clamped)
  {
    dim3 g((NI_ + 63) / 64, S_ / 64, B_);
    gemm_wmma<false, false, true, false><<<g, blk128, 0, stream>>>(
        Qb, Kb, nullptr, s2F, nullptr, S_, NI_, D_, D_, D_, NIP_,
        (long)S_ * D_, (long)NI_ * D_, (long)S_ * NIP_, iscale, NI_, NIP_);
  }
  softmax_cross_kernel<<<dim3((unsigned)MS), blk128, 0, stream>>>(s2F, probs2b);
  // attn2 = probs2 x V2  (K padded to 224; pad probs are zero; TDM-staged)
  {
    dim3 g(D_ / 128, S_ / 128, B_);
    gemm_wmma_tdm<false, false><<<g, blk256, 0, stream>>>(
        probs2b, Vb, nullptr, attn32, nullptr, S_, D_, NIP_, NIP_, D_, D_,
        (long)S_ * NIP_, (long)NI_ * D_, (long)S_ * D_, 1.0f);
  }
  add_ln_kernel<<<dim3((unsigned)MS), blk256, 0, stream>>>(attn32, x2f, g2, b2, x32, xb);

  // ---- vocab projection (dominant GEMM: 4096 x 32000 x 1024, TDM-staged) --
  {
    dim3 g(V_ / 128, (unsigned)(MS / 128), 1);
    gemm_wmma_tdm<true, false><<<g, blk256, 0, stream>>>(
        xb, Wpb, bp, out, nullptr, (int)MS, V_, D_, D_, V_, V_, 0, 0, 0, 1.0f);
  }
}
